// MEGNet_43379169689621
// MI455X (gfx1250) — matrix operand order repaired and checked by reading, not compiled, not run on hardware
//
#include <hip/hip_runtime.h>

#define LN2F 0.6931471805599453f

typedef __attribute__((ext_vector_type(16))) __bf16 v16bf;
typedef __attribute__((ext_vector_type(8)))  float  v8f;

__device__ __forceinline__ unsigned short f2b(float f) {
    unsigned u = __float_as_uint(f);
    return (unsigned short)((u + 0x7fffu + ((u >> 16) & 1u)) >> 16);  // RNE f32->bf16
}
__device__ __forceinline__ unsigned pack2bf(float a, float b) {
    return (unsigned)f2b(a) | ((unsigned)f2b(b) << 16);
}
__device__ __forceinline__ float softplus2f(float x) {
    return fmaxf(x, 0.f) + log1pf(__expf(-fabsf(x))) - LN2F;
}
__device__ __forceinline__ float sigf(float x) { return 1.f / (1.f + __expf(-x)); }
__device__ __forceinline__ unsigned fkey(float f) {
    unsigned u = __float_as_uint(f);
    return (u & 0x80000000u) ? ~u : (u | 0x80000000u);   // order-preserving float->uint
}
__device__ __forceinline__ float keyf(unsigned k) {
    return __uint_as_float((k & 0x80000000u) ? (k & 0x7fffffffu) : ~k);
}

struct Seg { const float* ptr; const int* idx; int w; };

struct MlpArgs {
    const float *W1, *B1, *W2, *B2, *W3, *B3;
    Seg seg[4]; int nseg; int kin;
    long nrows;
    float* out; const float* skip;
    float* atomA; const int* atomAidx;   // atomicAdd(atomA[idx[r]*OUT+n], y_pre_skip)
    float* atomB; const int* atomBidx;
};

// ---------------- WMMA layer helpers (CDNA5 bf16 16x16x32, f32 acc) ----------------
// A (16x32 bf16): lane = 16*hi + lo holds row M=lo; elem j -> K = hi*8 + (j<8 ? j : j+8)
//   -> per lane: two contiguous 8-elem runs => 2x ds_load_b128
// W stored TRANSPOSED in LDS: WT[n][k], row length K. B elem j -> K = hi*8 + (j<8 ? j : j+8), col = lo
//   -> per lane: two contiguous 8-elem runs => 2x ds_load_b128
// C/D (16x16 f32): reg r, lane -> M = r + 8*hi, N = lo

template<int K, int N>
__device__ __forceinline__ void wmma_layer_mid(const unsigned short* src, int ld,
                                               const unsigned short* WT, const float* B,
                                               unsigned short* dst, int m0, int hi, int lo) {
#pragma unroll
    for (int nt = 0; nt < N / 16; ++nt) {
        v8f acc; float bv = B[nt * 16 + lo];
#pragma unroll
        for (int r = 0; r < 8; ++r) acc[r] = bv;
        const unsigned short* wrow = WT + (nt * 16 + lo) * K + hi * 8;
        const unsigned short* arow = src + (m0 + lo) * ld + hi * 8;
#pragma unroll
        for (int kc = 0; kc < K / 32; ++kc) {
            union { v16bf v; uint4 q[2]; } af, bf;
            af.q[0] = *(const uint4*)(arow + kc * 32);
            af.q[1] = *(const uint4*)(arow + kc * 32 + 16);
            bf.q[0] = *(const uint4*)(wrow + kc * 32);
            bf.q[1] = *(const uint4*)(wrow + kc * 32 + 16);
            acc = __builtin_amdgcn_wmma_f32_16x16x32_bf16(false, af.v, false, bf.v,
                                                          (short)0, acc, false, false);
        }
#pragma unroll
        for (int r = 0; r < 8; ++r) {
            int row = m0 + r + 8 * hi;
            dst[row * 64 + nt * 16 + lo] = f2b(softplus2f(acc[r]));
        }
    }
}

template<int K, int N>
__device__ __forceinline__ void wmma_layer_fin(const unsigned short* src, int ld,
                                               const unsigned short* WT, const float* B,
                                               const MlpArgs& a, long r0, int m0, int hi, int lo) {
#pragma unroll
    for (int nt = 0; nt < N / 16; ++nt) {
        v8f acc; float bv = B[nt * 16 + lo];
#pragma unroll
        for (int r = 0; r < 8; ++r) acc[r] = bv;
        const unsigned short* wrow = WT + (nt * 16 + lo) * K + hi * 8;
        const unsigned short* arow = src + (m0 + lo) * ld + hi * 8;
#pragma unroll
        for (int kc = 0; kc < K / 32; ++kc) {
            union { v16bf v; uint4 q[2]; } af, bf;
            af.q[0] = *(const uint4*)(arow + kc * 32);
            af.q[1] = *(const uint4*)(arow + kc * 32 + 16);
            bf.q[0] = *(const uint4*)(wrow + kc * 32);
            bf.q[1] = *(const uint4*)(wrow + kc * 32 + 16);
            acc = __builtin_amdgcn_wmma_f32_16x16x32_bf16(false, af.v, false, bf.v,
                                                          (short)0, acc, false, false);
        }
        int n = nt * 16 + lo;
#pragma unroll
        for (int r = 0; r < 8; ++r) {
            long R = r0 + m0 + r + 8 * hi;
            if (R < a.nrows) {
                float y = softplus2f(acc[r]);
                if (a.atomA) atomicAdd(&a.atomA[(long)a.atomAidx[R] * N + n], y);
                if (a.atomB) atomicAdd(&a.atomB[(long)a.atomBidx[R] * N + n], y);
                float o = y; if (a.skip) o += a.skip[R * (long)N + n];
                a.out[R * (long)N + n] = o;
            }
        }
    }
}

// ---------------- Fused MLP kernel: 64 rows/block, 4 waves x 16-row tiles ----------------
template<int KP, int L1, int L2, int L3>   // L3==0 -> 2 layers (L2 = output)
__global__ __launch_bounds__(128) void fused_mlp_wmma(MlpArgs a) {
    constexpr int ROWS = 64;
    constexpr int W1N = L1 * KP, W2N = L2 * L1, W3N = (L3 > 0) ? L3 * L2 : 0;
    __shared__ unsigned short sA[ROWS * KP];
    __shared__ unsigned short sH[ROWS * 64];
    __shared__ unsigned short sH2[(L3 > 0) ? ROWS * 64 : 64];
    __shared__ unsigned short sW[W1N + W2N + ((W3N > 0) ? W3N : 1)];  // transposed [n][k]
    __shared__ float sB[L1 + L2 + ((L3 > 0) ? L3 : 1)];

    const int tid = threadIdx.x;
    // stage weights transposed (f32 -> bf16, packed pairs along k); pad k >= kin with zeros
    for (int i = tid; i < L1 * KP / 2; i += 128) {
        int nn = i / (KP / 2), k = (i % (KP / 2)) * 2;
        float v0 = (k < a.kin) ? a.W1[(long)k * L1 + nn] : 0.f;
        float v1 = (k + 1 < a.kin) ? a.W1[(long)(k + 1) * L1 + nn] : 0.f;
        *(unsigned*)&sW[nn * KP + k] = pack2bf(v0, v1);
    }
    for (int i = tid; i < L2 * L1 / 2; i += 128) {
        int nn = i / (L1 / 2), k = (i % (L1 / 2)) * 2;
        *(unsigned*)&sW[W1N + nn * L1 + k] =
            pack2bf(a.W2[(long)k * L2 + nn], a.W2[(long)(k + 1) * L2 + nn]);
    }
    if constexpr (L3 > 0) {
        for (int i = tid; i < L3 * L2 / 2; i += 128) {
            int nn = i / (L2 / 2), k = (i % (L2 / 2)) * 2;
            *(unsigned*)&sW[W1N + W2N + nn * L2 + k] =
                pack2bf(a.W3[(long)k * L3 + nn], a.W3[(long)(k + 1) * L3 + nn]);
        }
    }
    for (int n = tid; n < L1; n += 128) sB[n] = a.B1[n];
    for (int n = tid; n < L2; n += 128) sB[L1 + n] = a.B2[n];
    if constexpr (L3 > 0) {
        for (int n = tid; n < L3; n += 128) sB[L1 + L2 + n] = a.B3[n];
    }

    // stage gathered/concatenated input tile (pairs; all seg widths & kin are even)
    const long r0 = (long)blockIdx.x * ROWS;
    for (int i = tid; i < ROWS * KP / 2; i += 128) {
        int r = i / (KP / 2), kp = (i % (KP / 2)) * 2;
        float v0 = 0.f, v1 = 0.f; long R = r0 + r;
        if (R < a.nrows && kp < a.kin) {
            int kk = kp, s = 0;
            while (kk >= a.seg[s].w) { kk -= a.seg[s].w; ++s; }
            long row = a.seg[s].idx ? (long)a.seg[s].idx[R] : R;
            const float* p = &a.seg[s].ptr[row * (long)a.seg[s].w + kk];
            v0 = p[0]; v1 = p[1];
        }
        *(unsigned*)&sA[r * KP + kp] = pack2bf(v0, v1);
    }
    __syncthreads();

    const int wv = tid >> 5, ln = tid & 31, hi = ln >> 4, lo = ln & 15, m0 = wv * 16;

    wmma_layer_mid<KP, L1>(sA, KP, sW, sB, sH, m0, hi, lo);
    if constexpr (L3 > 0) {
        wmma_layer_mid<L1, L2>(sH, 64, sW + W1N, sB + L1, sH2, m0, hi, lo);
        wmma_layer_fin<L2, L3>(sH2, 64, sW + W1N + W2N, sB + L1 + L2, a, r0, m0, hi, lo);
    } else {
        wmma_layer_fin<L1, L2>(sH, 64, sW + W1N, sB + L1, a, r0, m0, hi, lo);
    }
}

// ---------------- Plain small-row MLP (per-graph work: G=128 rows) ----------------
struct SmallArgs {
    const float *W1, *B1, *W2, *B2, *W3, *B3;
    Seg seg[4]; int nseg; int kin;
    long nrows; float* out; const float* skip; int act_last;
};

template<int H1, int H2, int H3>
__global__ void small_mlp(SmallArgs a) {
    long r = (long)blockIdx.x * blockDim.x + threadIdx.x;
    if (r >= a.nrows) return;
    constexpr int OUT = (H3 > 0) ? H3 : H2;
    float h1[H1];
    for (int n = 0; n < H1; ++n) h1[n] = a.B1[n];
    {
        int s = 0, kk = 0;
        for (int k = 0; k < a.kin; ++k) {
            while (kk >= a.seg[s].w) { kk = 0; ++s; }
            long row = a.seg[s].idx ? (long)a.seg[s].idx[r] : r;
            float x = a.seg[s].ptr[row * (long)a.seg[s].w + kk]; ++kk;
            for (int n = 0; n < H1; ++n) h1[n] = fmaf(x, a.W1[k * H1 + n], h1[n]);
        }
    }
    for (int n = 0; n < H1; ++n) h1[n] = softplus2f(h1[n]);
    float h2[H2];
    for (int n = 0; n < H2; ++n) h2[n] = a.B2[n];
    for (int k = 0; k < H1; ++k)
        for (int n = 0; n < H2; ++n) h2[n] = fmaf(h1[k], a.W2[k * H2 + n], h2[n]);
    if constexpr (H3 > 0) {
        for (int n = 0; n < H2; ++n) h2[n] = softplus2f(h2[n]);
        float h3[(H3 > 0) ? H3 : 1];
        for (int n = 0; n < H3; ++n) h3[n] = a.B3[n];
        for (int k = 0; k < H2; ++k)
            for (int n = 0; n < H3; ++n) h3[n] = fmaf(h2[k], a.W3[k * H3 + n], h3[n]);
        for (int n = 0; n < H3; ++n) {
            float y = a.act_last ? softplus2f(h3[n]) : h3[n];
            if (a.skip) y += a.skip[r * (long)OUT + n];
            a.out[r * (long)OUT + n] = y;
        }
    } else {
        for (int n = 0; n < H2; ++n) {
            float y = a.act_last ? softplus2f(h2[n]) : h2[n];
            if (a.skip) y += a.skip[r * (long)OUT + n];
            a.out[r * (long)OUT + n] = y;
        }
    }
}

// ---------------- Utility / graph kernels ----------------
__global__ void zero_f32(float* p, long n) {
    long i = (long)blockIdx.x * blockDim.x + threadIdx.x;
    if (i < n) p[i] = 0.f;
}
__global__ void edge_prep(const int* src, const int* dst, const int* n2g,
                          int* e2g, float* ve_c, float* ue_c, long E) {
    long e = (long)blockIdx.x * blockDim.x + threadIdx.x;
    if (e >= E) return;
    int g = n2g[src[e]];
    e2g[e] = g;
    atomicAdd(&ve_c[dst[e]], 1.f);
    atomicAdd(&ue_c[g], 1.f);
}
__global__ void node_cnt(const int* n2g, float* uv_c, long V) {
    long v = (long)blockIdx.x * blockDim.x + threadIdx.x;
    if (v >= V) return;
    atomicAdd(&uv_c[n2g[v]], 1.f);
}
__global__ void finalize_mean(float* s, const float* c, long rows, int w) {
    long i = (long)blockIdx.x * blockDim.x + threadIdx.x;
    if (i >= rows * w) return;
    s[i] = s[i] / fmaxf(c[i / w], 1.f);
}

// ---------------- set2set ----------------
__global__ void lstm_step_k(const float* Wih, const float* Whh, const float* bih, const float* bhh,
                            const float* q, const float* hin, const float* cin,
                            float* hout, float* cout, int G) {
    int t = blockIdx.x * blockDim.x + threadIdx.x;
    if (t >= G * 32) return;
    int g = t >> 5, k = t & 31;
    float gi = bih[k]      + bhh[k];
    float gf = bih[32 + k] + bhh[32 + k];
    float gg = bih[64 + k] + bhh[64 + k];
    float go = bih[96 + k] + bhh[96 + k];
    const float* qq = q + g * 64;
    for (int j = 0; j < 64; ++j) {
        float x = qq[j];
        gi = fmaf(x, Wih[k * 64 + j], gi);
        gf = fmaf(x, Wih[(32 + k) * 64 + j], gf);
        gg = fmaf(x, Wih[(64 + k) * 64 + j], gg);
        go = fmaf(x, Wih[(96 + k) * 64 + j], go);
    }
    const float* hh = hin + g * 32;
    for (int j = 0; j < 32; ++j) {
        float x = hh[j];
        gi = fmaf(x, Whh[k * 32 + j], gi);
        gf = fmaf(x, Whh[(32 + k) * 32 + j], gf);
        gg = fmaf(x, Whh[(64 + k) * 32 + j], gg);
        go = fmaf(x, Whh[(96 + k) * 32 + j], go);
    }
    float cn = sigf(gf) * cin[g * 32 + k] + sigf(gi) * tanhf(gg);
    hout[g * 32 + k] = sigf(go) * tanhf(cn);
    cout[g * 32 + k] = cn;
}

// segment max with LDS-privatized bins (128 graphs), grid-stride
__global__ __launch_bounds__(256) void s2s_dot_max(const float* feat, const float* h, const int* ids,
                                                   float* esc, unsigned* mkey, long n) {
    __shared__ unsigned mk[128];
    for (int i = threadIdx.x; i < 128; i += 256) mk[i] = 0u;
    __syncthreads();
    for (long i = (long)blockIdx.x * 256 + threadIdx.x; i < n; i += (long)gridDim.x * 256) {
        int g = ids[i];
        const float* f = feat + i * 32; const float* hh = h + g * 32;
        float d = 0.f;
#pragma unroll
        for (int k = 0; k < 32; ++k) d = fmaf(f[k], hh[k], d);
        esc[i] = d;
        atomicMax(&mk[g], fkey(d));
    }
    __syncthreads();
    for (int i = threadIdx.x; i < 128; i += 256)
        if (mk[i]) atomicMax(&mkey[i], mk[i]);
}

// exp + weighted segment sums with LDS-privatized bins (128 graphs x 32), grid-stride
__global__ __launch_bounds__(256) void s2s_exp_sum(const float* feat, const float* esc, const int* ids,
                                                   const unsigned* mkey, float* ssum, float* rsum, long n) {
    __shared__ float bin[128 * 32];
    __shared__ float bs[128];
    for (int i = threadIdx.x; i < 128 * 32; i += 256) bin[i] = 0.f;
    for (int i = threadIdx.x; i < 128; i += 256) bs[i] = 0.f;
    __syncthreads();
    for (long i = (long)blockIdx.x * 256 + threadIdx.x; i < n; i += (long)gridDim.x * 256) {
        int g = ids[i];
        float ex = __expf(esc[i] - keyf(mkey[g]));
        atomicAdd(&bs[g], ex);
        const float* f = feat + i * 32;
#pragma unroll
        for (int k = 0; k < 32; ++k) atomicAdd(&bin[g * 32 + k], ex * f[k]);
    }
    __syncthreads();
    for (int i = threadIdx.x; i < 128 * 32; i += 256)
        if (bin[i] != 0.f) atomicAdd(&rsum[i], bin[i]);
    for (int i = threadIdx.x; i < 128; i += 256)
        if (bs[i] != 0.f) atomicAdd(&ssum[i], bs[i]);
}
__global__ void s2s_finish(const float* h, const float* rsum, const float* ssum, float* q, int G) {
    int t = blockIdx.x * blockDim.x + threadIdx.x;
    if (t >= G * 32) return;
    int g = t >> 5, k = t & 31;
    q[g * 64 + k]      = h[g * 32 + k];
    q[g * 64 + 32 + k] = rsum[g * 32 + k] / ssum[g];
}

// ---------------- Orchestration ----------------
extern "C" void kernel_launch(void* const* d_in, const int* in_sizes, int n_in,
                              void* d_out, int out_size, void* d_ws, size_t ws_size,
                              hipStream_t stream) {
    (void)in_sizes; (void)n_in; (void)out_size; (void)ws_size;
    const long E = 1600000, V = 50000; const int G = 128;

    int ii = 0;
    const float* edge_feat  = (const float*)d_in[ii++];
    const float* graph_attr = (const float*)d_in[ii++];
    const int*   node_feat  = (const int*)d_in[ii++];
    const int*   esrc       = (const int*)d_in[ii++];
    const int*   edst       = (const int*)d_in[ii++];
    const int*   n2g        = (const int*)d_in[ii++];
    auto F = [&]() { return (const float*)d_in[ii++]; };
    const float* node_emb = F();
    const float *neW1 = F(), *neB1 = F(), *neW2 = F(), *neB2 = F();   // node_enc
    const float *eeW1 = F(), *eeB1 = F(), *eeW2 = F(), *eeB2 = F();   // edge_enc
    const float *aeW1 = F(), *aeB1 = F(), *aeW2 = F(), *aeB2 = F();   // attr_enc
    struct Blk { bool dense; const float* d[12]; const float* ce[6]; const float* cn[6]; const float* ca[6]; } blk[3];
    for (int b = 0; b < 3; ++b) {
        blk[b].dense = (b > 0);
        if (b > 0) for (int j = 0; j < 12; ++j) blk[b].d[j] = F();  // dense_edge, dense_node, dense_attr (W,b,W,b each)
        for (int j = 0; j < 6; ++j) blk[b].ce[j] = F();
        for (int j = 0; j < 6; ++j) blk[b].cn[j] = F();
        for (int j = 0; j < 6; ++j) blk[b].ca[j] = F();
    }
    const float *nsWih = F(), *nsWhh = F(), *nsbih = F(), *nsbhh = F();
    const float *esWih = F(), *esWhh = F(), *esbih = F(), *esbhh = F();
    const float *oW1 = F(), *oB1 = F(), *oW2 = F(), *oB2 = F(), *oW3 = F(), *oB3 = F();

    // ---- workspace layout ----
    char* base = (char*)d_ws; size_t off = 0;
    auto A = [&](size_t nfl) -> float* {
        float* p = (float*)(base + off);
        off += (nfl * 4 + 255) & ~(size_t)255;
        return p;
    };
    float* ef   = A(E * 32); float* ef_t = A(E * 32);
    float* nf   = A(V * 32); float* nf_t = A(V * 32);
    float* u    = A(G * 32); float* u_t  = A(G * 32);
    int*   e2g  = (int*)A(E);
    float* ve_s = A(V * 32); float* ve_c = A(V);
    float* ue_s = A(G * 32); float* ue_c = A(G);
    float* uv_s = A(G * 32); float* uv_c = A(G);
    float* esc  = A(E);
    unsigned* mkey = (unsigned*)A(G);
    float* ssum = A(G); float* rsum = A(G * 32);
    float* qn = A(G * 64); float* qe = A(G * 64);
    float* hA = A(G * 32); float* hB = A(G * 32);
    float* cA = A(G * 32); float* cB = A(G * 32);

    auto zero = [&](float* p, long n) {
        zero_f32<<<dim3((unsigned)((n + 255) / 256)), 256, 0, stream>>>(p, n);
    };
    auto gsz = [](long n) {
        long b = (n + 255) / 256;
        return dim3((unsigned)(b < 1024 ? b : 1024));
    };

    // ---- graph structure prep ----
    zero(ve_c, V); zero(ue_c, G); zero(uv_c, G);
    edge_prep<<<dim3((unsigned)((E + 255) / 256)), 256, 0, stream>>>(esrc, edst, n2g, e2g, ve_c, ue_c, E);
    node_cnt<<<dim3((unsigned)((V + 255) / 256)), 256, 0, stream>>>(n2g, uv_c, V);

    // ---- encoders ----
    { // attr_enc: [G,2] -> 64 -> 32
        SmallArgs s{}; s.W1 = aeW1; s.B1 = aeB1; s.W2 = aeW2; s.B2 = aeB2;
        s.seg[0] = { graph_attr, nullptr, 2 }; s.nseg = 1; s.kin = 2;
        s.nrows = G; s.out = u; s.skip = nullptr; s.act_last = 1;
        small_mlp<64, 32, 0><<<1, 128, 0, stream>>>(s);
    }
    { // node_enc: emb gather [V,16] -> 64 -> 32 (WMMA)
        MlpArgs m{}; m.W1 = neW1; m.B1 = neB1; m.W2 = neW2; m.B2 = neB2;
        m.seg[0] = { node_emb, node_feat, 16 }; m.nseg = 1; m.kin = 16;
        m.nrows = V; m.out = nf;
        fused_mlp_wmma<32, 64, 32, 0><<<dim3((unsigned)((V + 63) / 64)), 128, 0, stream>>>(m);
    }
    { // edge_enc: [E,100] -> 64 -> 32 (WMMA)
        MlpArgs m{}; m.W1 = eeW1; m.B1 = eeB1; m.W2 = eeW2; m.B2 = eeB2;
        m.seg[0] = { edge_feat, nullptr, 100 }; m.nseg = 1; m.kin = 100;
        m.nrows = E; m.out = ef;
        fused_mlp_wmma<128, 64, 32, 0><<<dim3((unsigned)((E + 63) / 64)), 128, 0, stream>>>(m);
    }

    // ---- MEGNet blocks ----
    for (int b = 0; b < 3; ++b) {
        zero(ve_s, V * 32); zero(ue_s, G * 32); zero(uv_s, G * 32);
        const float *ef_in = ef, *nf_in = nf, *u_in = u;
        if (blk[b].dense) {
            { MlpArgs m{}; m.W1 = blk[b].d[0]; m.B1 = blk[b].d[1]; m.W2 = blk[b].d[2]; m.B2 = blk[b].d[3];
              m.seg[0] = { ef, nullptr, 32 }; m.nseg = 1; m.kin = 32; m.nrows = E; m.out = ef_t;
              fused_mlp_wmma<32, 64, 32, 0><<<dim3((unsigned)((E + 63) / 64)), 128, 0, stream>>>(m); }
            { MlpArgs m{}; m.W1 = blk[b].d[4]; m.B1 = blk[b].d[5]; m.W2 = blk[b].d[6]; m.B2 = blk[b].d[7];
              m.seg[0] = { nf, nullptr, 32 }; m.nseg = 1; m.kin = 32; m.nrows = V; m.out = nf_t;
              fused_mlp_wmma<32, 64, 32, 0><<<dim3((unsigned)((V + 63) / 64)), 128, 0, stream>>>(m); }
            { SmallArgs s{}; s.W1 = blk[b].d[8]; s.B1 = blk[b].d[9]; s.W2 = blk[b].d[10]; s.B2 = blk[b].d[11];
              s.seg[0] = { u, nullptr, 32 }; s.nseg = 1; s.kin = 32;
              s.nrows = G; s.out = u_t; s.skip = nullptr; s.act_last = 1;
              small_mlp<64, 32, 0><<<1, 128, 0, stream>>>(s); }
            ef_in = ef_t; nf_in = nf_t; u_in = u_t;
        }
        { // conv_edge: concat[nf[src],nf[dst],ef,u[e2g]] 128 -> 64 -> 64 -> 32, fused seg-sums
            MlpArgs m{};
            m.W1 = blk[b].ce[0]; m.B1 = blk[b].ce[1]; m.W2 = blk[b].ce[2]; m.B2 = blk[b].ce[3];
            m.W3 = blk[b].ce[4]; m.B3 = blk[b].ce[5];
            m.seg[0] = { nf_in, esrc, 32 }; m.seg[1] = { nf_in, edst, 32 };
            m.seg[2] = { ef_in, nullptr, 32 }; m.seg[3] = { u_in, e2g, 32 };
            m.nseg = 4; m.kin = 128; m.nrows = E;
            m.out = ef; m.skip = ef;                 // in-place skip (rows exclusive per block)
            m.atomA = ve_s; m.atomAidx = edst;       // ve sum over dst
            m.atomB = ue_s; m.atomBidx = e2g;        // ue sum over graph
            fused_mlp_wmma<128, 64, 64, 32><<<dim3((unsigned)((E + 63) / 64)), 128, 0, stream>>>(m);
        }
        finalize_mean<<<dim3((unsigned)((V * 32 + 255) / 256)), 256, 0, stream>>>(ve_s, ve_c, V, 32);
        { // conv_node: concat[nf,ve,u[n2g]] 96 -> 64 -> 64 -> 32, fused uv-sum
            MlpArgs m{};
            m.W1 = blk[b].cn[0]; m.B1 = blk[b].cn[1]; m.W2 = blk[b].cn[2]; m.B2 = blk[b].cn[3];
            m.W3 = blk[b].cn[4]; m.B3 = blk[b].cn[5];
            m.seg[0] = { nf_in, nullptr, 32 }; m.seg[1] = { ve_s, nullptr, 32 }; m.seg[2] = { u_in, n2g, 32 };
            m.nseg = 3; m.kin = 96; m.nrows = V;
            m.out = nf; m.skip = nf;
            m.atomA = uv_s; m.atomAidx = n2g;
            fused_mlp_wmma<96, 64, 64, 32><<<dim3((unsigned)((V + 63) / 64)), 128, 0, stream>>>(m);
        }
        finalize_mean<<<dim3((unsigned)((G * 32 + 255) / 256)), 256, 0, stream>>>(ue_s, ue_c, G, 32);
        finalize_mean<<<dim3((unsigned)((G * 32 + 255) / 256)), 256, 0, stream>>>(uv_s, uv_c, G, 32);
        { // conv_attr: concat[u,ue,uv] 96 -> 64 -> 64 -> 32 (+skip u)
            SmallArgs s{};
            s.W1 = blk[b].ca[0]; s.B1 = blk[b].ca[1]; s.W2 = blk[b].ca[2]; s.B2 = blk[b].ca[3];
            s.W3 = blk[b].ca[4]; s.B3 = blk[b].ca[5];
            s.seg[0] = { u_in, nullptr, 32 }; s.seg[1] = { ue_s, nullptr, 32 }; s.seg[2] = { uv_s, nullptr, 32 };
            s.nseg = 3; s.kin = 96; s.nrows = G; s.out = u; s.skip = u; s.act_last = 1;
            small_mlp<64, 64, 32><<<1, 128, 0, stream>>>(s);
        }
    }

    // ---- set2set pooling (2 iterations) ----
    auto set2set = [&](const float* feat, const int* ids, long n,
                       const float* Wih, const float* Whh, const float* bih, const float* bhh,
                       float* q) {
        zero(q, G * 64); zero(hA, G * 32); zero(cA, G * 32);
        float *hin = hA, *cin = cA, *hout = hB, *cout = cB;
        for (int it = 0; it < 2; ++it) {
            lstm_step_k<<<dim3((G * 32 + 255) / 256), 256, 0, stream>>>(Wih, Whh, bih, bhh, q, hin, cin, hout, cout, G);
            zero((float*)mkey, G);
            s2s_dot_max<<<gsz(n), 256, 0, stream>>>(feat, hout, ids, esc, mkey, n);
            zero(ssum, G); zero(rsum, G * 32);
            s2s_exp_sum<<<gsz(n), 256, 0, stream>>>(feat, esc, ids, mkey, ssum, rsum, n);
            s2s_finish<<<dim3((G * 32 + 255) / 256), 256, 0, stream>>>(hout, rsum, ssum, q, G);
            float* t;
            t = hin; hin = hout; hout = t;
            t = cin; cin = cout; cout = t;
        }
    };
    set2set(nf, n2g, V, nsWih, nsWhh, nsbih, nsbhh, qn);
    set2set(ef, e2g, E, esWih, esWhh, esbih, esbhh, qe);

    { // output head: concat[qn(64), qe(64), u(32)] = 160 -> 32 -> 16 -> 1 (no final act)
        SmallArgs s{};
        s.W1 = oW1; s.B1 = oB1; s.W2 = oW2; s.B2 = oB2; s.W3 = oW3; s.B3 = oB3;
        s.seg[0] = { qn, nullptr, 64 }; s.seg[1] = { qe, nullptr, 64 }; s.seg[2] = { u, nullptr, 32 };
        s.nseg = 3; s.kin = 160; s.nrows = G; s.out = (float*)d_out; s.skip = nullptr; s.act_last = 0;
        small_mlp<32, 16, 1><<<1, 128, 0, stream>>>(s);
    }
}
